// ISP_61641370632313
// MI455X (gfx1250) — matrix operand order repaired and checked by reading, not compiled, not run on hardware
//
#include <hip/hip_runtime.h>

#define N_   16
#define C_   3
#define H_   512
#define W_   512
#define HW_  (H_*W_)
#define BINS 2048
#define TOPK 13107          // int(512*512*0.05)
#define TR   64             // main-kernel tile rows
#define TC   32             // main-kernel tile cols

typedef __attribute__((ext_vector_type(16))) _Float16 v16h;
typedef __attribute__((ext_vector_type(8)))  float    v8f;

// ---- workspace layout (bytes) ----
static constexpr size_t OFF_TMPH = 0;                                   // N*HW f32 row box-sums
static constexpr size_t OFF_DARK = OFF_TMPH + (size_t)N_*HW_*4;         // N*HW f32 dark channel
static constexpr size_t OFF_HIST = OFF_DARK + (size_t)N_*HW_*4;         // N*BINS u32
static constexpr size_t OFF_TB   = OFF_HIST + (size_t)N_*BINS*4;        // 16 i32 threshold bin
static constexpr size_t OFF_FRAC = OFF_TB   + 64;                       // 16 f32 boundary frac
static constexpr size_t OFF_PART = OFF_FRAC + 64;                       // 16*64*3 f32 partial A sums
static constexpr size_t OFF_PAR  = OFF_PART + (size_t)N_*64*3*4;        // 16*16 f32 squashed params
static constexpr size_t OFF_MN   = OFF_PAR  + (size_t)N_*16*4;          // 48 u32 per-(n,c) min bits
static constexpr size_t OFF_MX   = OFF_MN   + 192;                      // 48 u32 per-(n,c) max bits

__device__ __forceinline__ void gauss5(float g[5]) {
  float p0 = __expf(-2.0f), p1 = __expf(-0.5f);
  float s  = 1.0f + 2.0f*p1 + 2.0f*p0 + 0.001f;   // matches phi.sum()+0.001
  g[0] = p0/s; g[1] = p1/s; g[2] = 1.0f/s; g[3] = g[1]; g[4] = g[0];
}

// ---- 0: init scratch that needs defined initial values ----
__global__ void k_init(unsigned int* __restrict__ hist, unsigned int* __restrict__ mn,
                       unsigned int* __restrict__ mx) {
  int i = blockIdx.x*blockDim.x + threadIdx.x;
  if (i < N_*BINS) hist[i] = 0u;
  if (i < N_*C_) { mn[i] = 0x7F800000u; mx[i] = 0u; }   // +inf / 0 (x_sum > 0)
}

// ---- 1: channel-min + horizontal 15-wide box sum (zero padded) ----
__global__ void __launch_bounds__(256) k_minrow(const float* __restrict__ x,
                                                float* __restrict__ tmpH) {
  int row = blockIdx.x % H_;
  int n   = blockIdx.x / H_;
  __shared__ float m[W_ + 14];
  const float* x0 = x + (size_t)n*C_*HW_ + (size_t)row*W_;
  for (int i = threadIdx.x; i < W_ + 14; i += blockDim.x) {
    int c = i - 7;
    float v = 0.0f;
    if (c >= 0 && c < W_) {
      float a = x0[c], b = x0[HW_ + c], d = x0[2*HW_ + c];
      v = fminf(a, fminf(b, d));
    }
    m[i] = v;
  }
  __syncthreads();
  float* out = tmpH + (size_t)n*HW_ + (size_t)row*W_;
  for (int c = threadIdx.x; c < W_; c += blockDim.x) {
    float s = 0.0f;
    #pragma unroll
    for (int d = 0; d < 15; ++d) s += m[c + d];
    out[c] = s;
  }
}

// ---- 2: vertical 15-wide sliding sum -> dark = sum/225 ----
__global__ void __launch_bounds__(256) k_darkcol(const float* __restrict__ tmpH,
                                                 float* __restrict__ dark) {
  int n = blockIdx.x >> 1;
  int c = ((blockIdx.x & 1) << 8) + threadIdx.x;
  const float* T = tmpH + (size_t)n*HW_ + c;
  float*       D = dark + (size_t)n*HW_ + c;
  float s = 0.0f;
  for (int r = 0; r < 7; ++r) s += T[r*W_];
  for (int r = 0; r < H_; ++r) {
    int ra = r + 7;
    if (ra < H_) s += T[ra*W_];
    D[r*W_] = s * (1.0f/225.0f);
    int rb = r - 7;
    if (rb >= 0) s -= T[rb*W_];
  }
}

// ---- 3: per-sample histogram of dark ----
__global__ void __launch_bounds__(256) k_hist(const float* __restrict__ dark,
                                              unsigned int* __restrict__ hist) {
  int n = blockIdx.x >> 4, chunk = blockIdx.x & 15;
  __shared__ unsigned int h[BINS];
  for (int i = threadIdx.x; i < BINS; i += blockDim.x) h[i] = 0u;
  __syncthreads();
  const float* D = dark + (size_t)n*HW_ + (size_t)chunk*(HW_/16);
  for (int i = threadIdx.x; i < HW_/16; i += blockDim.x) {
    int b = (int)(D[i] * (float)BINS);
    b = b < 0 ? 0 : (b > BINS-1 ? BINS-1 : b);
    atomicAdd(&h[b], 1u);
  }
  __syncthreads();
  unsigned int* Hg = hist + n*BINS;
  for (int i = threadIdx.x; i < BINS; i += blockDim.x)
    if (h[i]) atomicAdd(&Hg[i], h[i]);
}

// ---- 4: find top-5% threshold bin + fractional boundary weight ----
__global__ void k_thresh(const unsigned int* __restrict__ hist,
                         int* __restrict__ tb, float* __restrict__ frac) {
  int n = threadIdx.x;
  if (n >= N_) return;
  const unsigned int* Hh = hist + n*BINS;
  unsigned int cum = 0; int b = 0; float f = 1.0f;
  for (int i = BINS-1; i >= 0; --i) {
    unsigned int c = Hh[i];
    if (cum + c >= (unsigned)TOPK) {
      b = i; f = (float)(TOPK - cum) / (float)(c > 0u ? c : 1u); break;
    }
    cum += c;
  }
  tb[n] = b; frac[n] = f;
}

// ---- 5: weighted sums of x over top-dark pixels (deterministic partials) ----
__global__ void __launch_bounds__(256) k_suma(const float* __restrict__ x,
        const float* __restrict__ dark, const int* __restrict__ tb,
        const float* __restrict__ frac, float* __restrict__ part) {
  int n = blockIdx.x >> 6, chunk = blockIdx.x & 63;
  int base = chunk * (HW_/64);
  int t = tb[n]; float f = frac[n];
  const float* D = dark + (size_t)n*HW_ + base;
  const float* X = x + (size_t)n*C_*HW_ + base;
  float s0 = 0.f, s1 = 0.f, s2 = 0.f;
  for (int i = threadIdx.x; i < HW_/64; i += blockDim.x) {
    int b = (int)(D[i] * (float)BINS);
    b = b < 0 ? 0 : (b > BINS-1 ? BINS-1 : b);
    float wgt = (b > t) ? 1.0f : ((b == t) ? f : 0.0f);
    if (wgt > 0.0f) {
      s0 += wgt * X[i]; s1 += wgt * X[HW_ + i]; s2 += wgt * X[2*HW_ + i];
    }
  }
  #pragma unroll
  for (int off = 16; off > 0; off >>= 1) {
    s0 += __shfl_xor(s0, off, 32);
    s1 += __shfl_xor(s1, off, 32);
    s2 += __shfl_xor(s2, off, 32);
  }
  __shared__ float r[3][8];
  int wid = threadIdx.x >> 5, lane = threadIdx.x & 31;
  if (lane == 0) { r[0][wid] = s0; r[1][wid] = s1; r[2][wid] = s2; }
  __syncthreads();
  if (threadIdx.x == 0) {
    float a0 = 0.f, a1 = 0.f, a2 = 0.f;
    for (int k = 0; k < 8; ++k) { a0 += r[0][k]; a1 += r[1][k]; a2 += r[2][k]; }
    float* p = part + ((size_t)n*64 + chunk)*3;
    p[0] = a0; p[1] = a1; p[2] = a2;
  }
}

// ---- 6: parameter squashing + atmospheric light A ----
__global__ void k_params(const float* __restrict__ o1raw, const float* __restrict__ o2raw,
                         const float* __restrict__ part, float* __restrict__ P) {
  int n = threadIdx.x;
  if (n >= N_) return;
  auto t01 = [](float v) { return tanhf(v)*0.5f + 0.5f; };
  const float* a = o1raw + n*7;
  float param = t01(a[0])*0.25f + 1.0f;
  float lamda = t01(a[2])*2.0f;
  float w     = t01(a[3]);
  const float* b = o2raw + n*4;
  float v0 = t01(b[0]), v1 = t01(b[1]), v2 = t01(b[2]), v3 = t01(b[3]);
  float mxv = fmaxf(fmaxf(v0, v1), fmaxf(v2, v3));
  float e0 = __expf(v0-mxv), e1 = __expf(v1-mxv), e2 = __expf(v2-mxv), e3 = __expf(v3-mxv);
  float es = e0+e1+e2+e3;
  float o20 = e0/es, o21 = e1/es, o22 = e2/es, o23 = e3/es;
  float* p = P + n*16;
  p[0] = param; p[1] = lamda; p[2] = w;
  p[3] = t01(a[4])*0.1f + 0.95f;
  p[4] = t01(a[5])*0.1f + 0.95f;
  p[5] = t01(a[6])*0.1f + 0.95f;
  p[6] = o20; p[7] = o21; p[8] = o22; p[9] = o23;
  p[10] = 1.0f / (o20+o21+o22+o23 + 0.001f);
  float A0 = 0.f, A1 = 0.f, A2 = 0.f;
  for (int k = 0; k < 64; ++k) {                 // fixed order -> deterministic
    const float* q = part + ((size_t)n*64 + k)*3;
    A0 += q[0]; A1 += q[1]; A2 += q[2];
  }
  p[11] = A0/(float)TOPK; p[12] = A1/(float)TOPK; p[13] = A2/(float)TOPK;
}

// ---- 7: fused main: blur (horiz VALU + vert WMMA) + all branches + x_sum + min/max ----
__global__ void __launch_bounds__(256) k_main(const float* __restrict__ x,
        const float* __restrict__ dark, const float* __restrict__ P,
        float* __restrict__ out, unsigned int* __restrict__ mn, unsigned int* __restrict__ mx) {
  int nc = blockIdx.z;
  int n = nc / C_, c = nc % C_;
  int r0 = blockIdx.y * TR, c0 = blockIdx.x * TC;

  __shared__ float    sx[TR+4][TC+5];    // x tile rows -2..65, cols -2..33 (+1 pad)
  __shared__ _Float16 sh[TR+4][TC];      // horizontally blurred (rows -2..65)
  __shared__ float    sb[TR][TC];        // 2D blurred

  const float* X = x + (size_t)nc*HW_;
  for (int i = threadIdx.x; i < (TR+4)*(TC+4); i += blockDim.x) {
    int rr = i / (TC+4), cc = i % (TC+4);
    int gr = r0 + rr - 2, gc = c0 + cc - 2;
    float v = 0.0f;                                      // conv zero padding
    if (gr >= 0 && gr < H_ && gc >= 0 && gc < W_) v = X[(size_t)gr*W_ + gc];
    sx[rr][cc] = v;
  }
  __syncthreads();

  float g[5]; gauss5(g);

  for (int i = threadIdx.x; i < (TR+4)*TC; i += blockDim.x) {
    int rr = i / TC, cc = i % TC;
    float s = g[0]*sx[rr][cc]   + g[1]*sx[rr][cc+1] + g[2]*sx[rr][cc+2]
            + g[3]*sx[rr][cc+3] + g[4]*sx[rr][cc+4];
    sh[rr][cc] = (_Float16)s;
  }
  __syncthreads();

  // Vertical 5-tap as banded matmul: D[16,16] = Aband[16,32(K, 20 used)] x Hext[32,16].
  // One 16x16 subtile per wave (8 waves = 4x2 subtiles of the 64x32 tile).
  {
    int wid  = threadIdx.x >> 5;
    int lane = threadIdx.x & 31;
    int sr = wid >> 1, sc = wid & 1;
    int m = lane & 15, half = lane >> 4;
    v16h a, b;
    #pragma unroll
    for (int e = 0; e < 16; ++e) {
      // 16-bit A layout: lane half selects K+8; element e -> K = 16*(e>>3) + 8*half + (e&7)
      int K = 16*(e >> 3) + 8*half + (e & 7);
      int d = K - m;                                    // band: A[m][K] = g[K-m]
      a[e] = (_Float16)((K < 20 && d >= 0 && d < 5) ? g[d] : 0.0f);
      // 16-bit B (32x16) layout: same K order as A: K = 16*(e>>3) + 8*half + (e&7); col = lane&15
      int j = 16*(e >> 3) + 8*half + (e & 7);           // extended row index 0..31
      b[e] = (j < 20) ? sh[sr*16 + j][sc*16 + m] : (_Float16)0.0f;
    }
    v8f acc = {};
    acc = __builtin_amdgcn_wmma_f32_16x16x32_f16(false, a, false, b, (short)0, acc,
                                                 false, false);
    #pragma unroll
    for (int p2 = 0; p2 < 8; ++p2)
      sb[sr*16 + 8*half + p2][sc*16 + m] = acc[p2];     // D: VGPR p -> row p+8*half
  }
  __syncthreads();

  const float* p = P + n*16;
  float param = p[0], lamda = p[1], w = p[2];
  float wbc = p[3 + c];
  float w0 = p[6], w1 = p[7], w2 = p[8], w3 = p[9], inv_den = p[10];
  float A = p[11 + c];

  const float* D = dark + (size_t)n*HW_;
  float* O = out + (size_t)nc*HW_;

  float lmn = 3.4e38f, lmx = 0.0f;
  for (int i = threadIdx.x; i < TR*TC; i += blockDim.x) {
    int rr = i >> 5, cc = i & 31;
    float xv = sx[rr+2][cc+2];
    float x1 = fminf(fmaxf(xv*wbc, 0.0f), 1.0f);                       // WB
    float x2 = fminf(fmaxf(__powf(xv, param) + 0.001f, 0.0f), 1.0f);   // gamma
    float bl = fminf(fmaxf(sb[rr][cc], 0.0f), 1.0f);                   // blurred
    float x4 = fminf(fmaxf(xv + lamda*(xv - bl), 0.0f), 1.0f);         // sharpen
    float dk = D[(size_t)(r0+rr)*W_ + (c0+cc)];
    float tr = 1.0f - w*dk; tr = (tr > 0.1f) ? tr : 0.1f;
    float x5 = fminf(fmaxf((xv - A)/(tr + 0.001f) + A, 0.0f), 1.0f);   // dehaze
    float xs = (xv + 0.1f*(w0*x1 + w1*x2 + w2*x4 + w3*x5)) * inv_den;
    O[(size_t)(r0+rr)*W_ + (c0+cc)] = xs;
    lmn = fminf(lmn, xs); lmx = fmaxf(lmx, xs);
  }
  #pragma unroll
  for (int off = 16; off > 0; off >>= 1) {
    lmn = fminf(lmn, __shfl_xor(lmn, off, 32));
    lmx = fmaxf(lmx, __shfl_xor(lmx, off, 32));
  }
  if ((threadIdx.x & 31) == 0) {           // positive floats: uint order == float order
    atomicMin(&mn[nc], __float_as_uint(lmn));
    atomicMax(&mx[nc], __float_as_uint(lmx));
  }
}

// ---- 8: in-place per-(n,c) min-max normalization ----
__global__ void __launch_bounds__(256) k_norm(float* __restrict__ out,
        const unsigned int* __restrict__ mn, const unsigned int* __restrict__ mx) {
  size_t idx = (size_t)blockIdx.x * blockDim.x + threadIdx.x;
  if (idx >= (size_t)N_*C_*HW_) return;
  int nc = (int)(idx / HW_);
  float a = __uint_as_float(mn[nc]);
  float b = __uint_as_float(mx[nc]);
  out[idx] = (out[idx] - a) / (b - a + 1e-7f);
}

extern "C" void kernel_launch(void* const* d_in, const int* in_sizes, int n_in,
                              void* d_out, int out_size, void* d_ws, size_t ws_size,
                              hipStream_t stream) {
  (void)in_sizes; (void)n_in; (void)out_size; (void)ws_size;
  const float* x  = (const float*)d_in[0];
  const float* o1 = (const float*)d_in[1];
  const float* o2 = (const float*)d_in[2];
  float* out = (float*)d_out;
  char* ws = (char*)d_ws;
  float*        tmpH = (float*)(ws + OFF_TMPH);
  float*        dark = (float*)(ws + OFF_DARK);
  unsigned int* hist = (unsigned int*)(ws + OFF_HIST);
  int*          tb   = (int*)(ws + OFF_TB);
  float*        frac = (float*)(ws + OFF_FRAC);
  float*        part = (float*)(ws + OFF_PART);
  float*        P    = (float*)(ws + OFF_PAR);
  unsigned int* mn   = (unsigned int*)(ws + OFF_MN);
  unsigned int* mx   = (unsigned int*)(ws + OFF_MX);

  k_init   <<<(N_*BINS + 255)/256, 256, 0, stream>>>(hist, mn, mx);
  k_minrow <<<N_*H_,               256, 0, stream>>>(x, tmpH);
  k_darkcol<<<N_*(W_/256),         256, 0, stream>>>(tmpH, dark);
  k_hist   <<<N_*16,               256, 0, stream>>>(dark, hist);
  k_thresh <<<1,                    32, 0, stream>>>(hist, tb, frac);
  k_suma   <<<N_*64,               256, 0, stream>>>(x, dark, tb, frac, part);
  k_params <<<1,                    32, 0, stream>>>(o1, o2, part, P);
  dim3 grid(W_/TC, H_/TR, N_*C_);
  k_main   <<<grid,                256, 0, stream>>>(x, dark, P, out, mn, mx);
  size_t total = (size_t)N_*C_*HW_;
  k_norm   <<<(unsigned)((total + 255)/256), 256, 0, stream>>>(out, mn, mx);
}
